// MultiheadAttention_28733331210495
// MI455X (gfx1250) — matrix operand order repaired and checked by reading, not compiled, not run on hardware
//
#include <hip/hip_runtime.h>
#include <hip/hip_bf16.h>

typedef __attribute__((ext_vector_type(16))) _Float16 v16h;
typedef __attribute__((ext_vector_type(8)))  _Float16 v8h;
typedef __attribute__((ext_vector_type(4)))  _Float16 v4h;
typedef __attribute__((ext_vector_type(8)))  float    v8f;

#define WMMA_F16(A_, B_, C_) \
  __builtin_amdgcn_wmma_f32_16x16x32_f16(false, (A_), false, (B_), (short)0, (C_), false, false)

// ---------------- fragment loaders (per CDNA5 ISA layouts) ----------------

// A matrix 16x32 f16: lane r = L&15, VGPR v holds K = (v>=4?16:0) + 8*(L>>4) + 2*(v&3)
__device__ __forceinline__ v16h load_frag_a(const _Float16* __restrict__ p, int ld) {
  const int lane = threadIdx.x & 31;
  const int r = lane & 15;
  const int h = lane >> 4;
  const _Float16* q = p + (size_t)r * ld + h * 8;
  v8h lo = *(const v8h*)(q);        // K = h*8 .. h*8+7   -> VGPRs 0..3
  v8h hi = *(const v8h*)(q + 16);   // K = 16+h*8 .. +7   -> VGPRs 4..7
  v16h f;
#pragma unroll
  for (int i = 0; i < 8; ++i) { f[i] = lo[i]; f[8 + i] = hi[i]; }
  return f;
}

// B matrix 32x16 f16 loaded from B^T (row-major N x K): lane n = L&15,
// VGPR v holds K = (v>=4?16:0) + 8*(L>>4) + 2*(v&3)
__device__ __forceinline__ v16h load_frag_b(const _Float16* __restrict__ p, int ld) {
  const int lane = threadIdx.x & 31;
  const int n = lane & 15;
  const int h = lane >> 4;
  const _Float16* q = p + (size_t)n * ld + h * 8;
  v8h lo = *(const v8h*)(q);
  v8h hi = *(const v8h*)(q + 16);
  v16h f;
#pragma unroll
  for (int i = 0; i < 8; ++i) { f[i] = lo[i]; f[8 + i] = hi[i]; }
  return f;
}

// C/D 16x16 f32: lane n = L&15, VGPR v is row m = v + 8*(L>>4)
__device__ __forceinline__ void store_tile_f32(float* __restrict__ C, int ldc, v8f acc) {
  const int lane = threadIdx.x & 31;
  const int n = lane & 15;
  const int h = lane >> 4;
#pragma unroll
  for (int v = 0; v < 8; ++v) C[(size_t)(v + 8 * h) * ldc + n] = acc[v];
}

__device__ __forceinline__ float rowred_max(float x) {
#pragma unroll
  for (int m = 1; m <= 8; m <<= 1) x = fmaxf(x, __shfl_xor(x, m, 32));
  return x;
}
__device__ __forceinline__ float rowred_sum(float x) {
#pragma unroll
  for (int m = 1; m <= 8; m <<= 1) x += __shfl_xor(x, m, 32);
  return x;
}

// ---------------- elementwise / conversion kernels ----------------

__global__ void f32_to_f16_kernel(const float* __restrict__ in, _Float16* __restrict__ out, int nquad) {
  int i = blockIdx.x * blockDim.x + threadIdx.x;
  if (i >= nquad) return;
  float4 v = ((const float4*)in)[i];
  v4h o;
  o[0] = (_Float16)v.x; o[1] = (_Float16)v.y; o[2] = (_Float16)v.z; o[3] = (_Float16)v.w;
  ((v4h*)out)[i] = o;
}

// RoPE on Q part of qkv (fp32, 2048 x 6144), write f16 qh (2048 x 4096)
__global__ void rope_q_kernel(const float* __restrict__ qkv, const float* __restrict__ fc,
                              _Float16* __restrict__ qh) {
  int i = blockIdx.x * blockDim.x + threadIdx.x;   // over 2048*32*64
  if (i >= 2048 * 32 * 64) return;
  int p  = i & 63;
  int hh = (i >> 6) & 31;
  int s  = i >> 11;
  const float* base = qkv + (size_t)s * 6144 + hh * 128 + p * 2;
  float c  = fc[s * 128 + p * 2];
  float sn = fc[s * 128 + p * 2 + 1];
  float t0 = base[0], t1 = base[1];
  _Float16* ob = qh + (size_t)s * 4096 + hh * 128 + p * 2;
  ob[0] = (_Float16)(t0 * c - t1 * sn);
  ob[1] = (_Float16)(t0 * sn + t1 * c);
}

// RoPE on K part of qkv, write f16 kh (2048 x 1024)
__global__ void rope_k_kernel(const float* __restrict__ qkv, const float* __restrict__ fc,
                              _Float16* __restrict__ kh) {
  int i = blockIdx.x * blockDim.x + threadIdx.x;   // over 2048*8*64
  if (i >= 2048 * 8 * 64) return;
  int p  = i & 63;
  int hh = (i >> 6) & 7;
  int s  = i >> 9;
  const float* base = qkv + (size_t)s * 6144 + 4096 + hh * 128 + p * 2;
  float c  = fc[s * 128 + p * 2];
  float sn = fc[s * 128 + p * 2 + 1];
  float t0 = base[0], t1 = base[1];
  _Float16* ob = kh + (size_t)s * 1024 + hh * 128 + p * 2;
  ob[0] = (_Float16)(t0 * c - t1 * sn);
  ob[1] = (_Float16)(t0 * sn + t1 * c);
}

// V part of qkv (2048 x 1024 at col offset 5120) -> vt f16 (8 x 128 x 2048): vt[h][d][s]
__global__ void transpose_v_kernel(const float* __restrict__ qkv, _Float16* __restrict__ vt) {
  int i = blockIdx.x * blockDim.x + threadIdx.x;   // over 8*128*2048
  if (i >= 8 * 128 * 2048) return;
  int s  = i & 2047;
  int d  = (i >> 11) & 127;
  int hh = i >> 18;
  vt[((size_t)hh * 128 + d) * 2048 + s] =
      (_Float16)qkv[(size_t)s * 6144 + 5120 + hh * 128 + d];
}

// ---------------- GEMM: C(MxN,f32) = A(MxK,f16,row) * B(NxK,f16,row)^T ----------------
// block = 256 threads (8 waves, 2Mx4N), block tile 64x256, wave tile 32x64.
__global__ __launch_bounds__(256) void gemm_f16_nt_kernel(
    const _Float16* __restrict__ A, const _Float16* __restrict__ B,
    float* __restrict__ C, int K, int lda, int ldb, int ldc)
{
  const int wave = threadIdx.x >> 5;
  const int wm = wave >> 2;
  const int wn = wave & 3;
  const int m0 = blockIdx.y * 64 + wm * 32;
  const int n0 = blockIdx.x * 256 + wn * 64;

  v8f acc[2][4] = {};
  for (int k0 = 0; k0 < K; k0 += 32) {
    v16h a0 = load_frag_a(A + (size_t)m0 * lda + k0, lda);
    v16h a1 = load_frag_a(A + (size_t)(m0 + 16) * lda + k0, lda);
    v16h b0 = load_frag_b(B + (size_t)(n0 +  0) * ldb + k0, ldb);
    v16h b1 = load_frag_b(B + (size_t)(n0 + 16) * ldb + k0, ldb);
    v16h b2 = load_frag_b(B + (size_t)(n0 + 32) * ldb + k0, ldb);
    v16h b3 = load_frag_b(B + (size_t)(n0 + 48) * ldb + k0, ldb);
    acc[0][0] = WMMA_F16(a0, b0, acc[0][0]);
    acc[0][1] = WMMA_F16(a0, b1, acc[0][1]);
    acc[0][2] = WMMA_F16(a0, b2, acc[0][2]);
    acc[0][3] = WMMA_F16(a0, b3, acc[0][3]);
    acc[1][0] = WMMA_F16(a1, b0, acc[1][0]);
    acc[1][1] = WMMA_F16(a1, b1, acc[1][1]);
    acc[1][2] = WMMA_F16(a1, b2, acc[1][2]);
    acc[1][3] = WMMA_F16(a1, b3, acc[1][3]);
  }
#pragma unroll
  for (int i = 0; i < 2; ++i)
#pragma unroll
    for (int j = 0; j < 4; ++j)
      store_tile_f32(C + (size_t)(m0 + i * 16) * ldc + (n0 + j * 16), ldc, acc[i][j]);
}

// ---------------- Flash attention: one wave = one 16-query tile of one head -------------
// qh: (2048 x 4096) f16 (RoPE'd), kh: (2048 x 1024) f16 (RoPE'd),
// vt: (8 x 128 x 2048) f16, yh out: (2048 x 4096) f16
__global__ __launch_bounds__(256) void attn_kernel(
    const _Float16* __restrict__ qh, const _Float16* __restrict__ kh,
    const _Float16* __restrict__ vt, _Float16* __restrict__ yh)
{
  __shared__ __align__(16) _Float16 plds[8][16 * 32];   // 1 KB per wave
  const int wave = threadIdx.x >> 5;
  const int lane = threadIdx.x & 31;
  const int n = lane & 15;
  const int h = lane >> 4;

  const int idx   = blockIdx.x * 8 + wave;   // 0..4095
  const int qt    = idx & 127;               // query tile
  const int hh    = idx >> 7;                // head 0..31
  const int kvh   = hh >> 2;                 // GQA: 4 Q heads per KV head
  const int qBase = qt * 16;
  const float scale = 0.08838834764831845f;  // 1/sqrt(128)

  v16h qf[4];
#pragma unroll
  for (int ks = 0; ks < 4; ++ks)
    qf[ks] = load_frag_a(qh + (size_t)qBase * 4096 + hh * 128 + ks * 32, 4096);

  v8f o[8] = {};
  float mrow[8], lrow[8];
#pragma unroll
  for (int v = 0; v < 8; ++v) { mrow[v] = -1e30f; lrow[v] = 0.0f; }

  for (int kb = 0; kb < qBase + 16; kb += 32) {
    // S = Q K^T for 32 keys: two 16x16 tiles over 4 dh chunks of 32
    v8f s0 = {}, s1 = {};
#pragma unroll
    for (int ks = 0; ks < 4; ++ks) {
      v16h bk0 = load_frag_b(kh + (size_t)(kb)      * 1024 + kvh * 128 + ks * 32, 1024);
      v16h bk1 = load_frag_b(kh + (size_t)(kb + 16) * 1024 + kvh * 128 + ks * 32, 1024);
      s0 = WMMA_F16(qf[ks], bk0, s0);
      s1 = WMMA_F16(qf[ks], bk1, s1);
    }
    // scale + causal mask (element: row m = v+8h -> query qBase+m; col n -> key)
    float sv[2][8];
#pragma unroll
    for (int v = 0; v < 8; ++v) {
      int qi = qBase + v + 8 * h;
      sv[0][v] = ((kb + n)      <= qi) ? s0[v] * scale : -1e30f;
      sv[1][v] = ((kb + 16 + n) <= qi) ? s1[v] * scale : -1e30f;
    }
    // online softmax per row
#pragma unroll
    for (int v = 0; v < 8; ++v) {
      float rm   = rowred_max(fmaxf(sv[0][v], sv[1][v]));
      float newm = fmaxf(mrow[v], rm);
      float fac  = __expf(mrow[v] - newm);
      mrow[v] = newm;
      float p0 = __expf(sv[0][v] - newm);
      float p1 = __expf(sv[1][v] - newm);
      lrow[v] = lrow[v] * fac + rowred_sum(p0 + p1);
#pragma unroll
      for (int d = 0; d < 8; ++d) o[d][v] *= fac;
      sv[0][v] = p0; sv[1][v] = p1;
    }
    // stage P (16x32 f16) through LDS to convert C-layout -> A-layout
    _Float16* pl = plds[wave];
#pragma unroll
    for (int v = 0; v < 8; ++v) {
      pl[(v + 8 * h) * 32 + n]      = (_Float16)sv[0][v];
      pl[(v + 8 * h) * 32 + 16 + n] = (_Float16)sv[1][v];
    }
    v16h pf = load_frag_a(pl, 32);
    // O += P V  (B^T = vt is (dh x key), key-contiguous)
#pragma unroll
    for (int d = 0; d < 8; ++d) {
      v16h bv = load_frag_b(vt + ((size_t)kvh * 128 + d * 16) * 2048 + kb, 2048);
      o[d] = WMMA_F16(pf, bv, o[d]);
    }
  }

  float inv[8];
#pragma unroll
  for (int v = 0; v < 8; ++v) inv[v] = 1.0f / lrow[v];
#pragma unroll
  for (int d = 0; d < 8; ++d)
#pragma unroll
    for (int v = 0; v < 8; ++v)
      yh[(size_t)(qBase + v + 8 * h) * 4096 + hh * 128 + d * 16 + n] =
          (_Float16)(o[d][v] * inv[v]);
}

// ---------------- launch ----------------

extern "C" void kernel_launch(void* const* d_in, const int* in_sizes, int n_in,
                              void* d_out, int out_size, void* d_ws, size_t ws_size,
                              hipStream_t stream) {
  (void)in_sizes; (void)n_in; (void)out_size; (void)ws_size;
  const float* x    = (const float*)d_in[0];   // 2048 x 4096
  const float* fc   = (const float*)d_in[1];   // 2048 x 64 x 2
  const float* Wqkv = (const float*)d_in[2];   // 6144 x 4096
  const float* Wo   = (const float*)d_in[3];   // 4096 x 4096
  float* out = (float*)d_out;                  // 2048 x 4096

  char* ws = (char*)d_ws;
  const size_t MB = 1024ull * 1024ull;
  float*    qkv = (float*)(ws);                 // 48 MB  fp32 QKV projections
  _Float16* qh  = (_Float16*)(ws + 48 * MB);    // 16 MB  RoPE'd Q (f16)
  _Float16* kh  = (_Float16*)(ws + 64 * MB);    //  4 MB  RoPE'd K (f16)
  _Float16* vt  = (_Float16*)(ws + 68 * MB);    //  4 MB  V transposed (f16)
  _Float16* xh  = (_Float16*)(ws + 72 * MB);    // 16 MB  x f16, later reused as attn out
  _Float16* wh  = (_Float16*)(ws + 88 * MB);    // 48 MB  weights f16 (Wqkv then Wo)

  const int T = 256;
  // convert x and W_qkv to f16
  f32_to_f16_kernel<<<(2048 * 4096 / 4 + T - 1) / T, T, 0, stream>>>(x, xh, 2048 * 4096 / 4);
  f32_to_f16_kernel<<<(6144 * 4096 / 4 + T - 1) / T, T, 0, stream>>>(Wqkv, wh, 6144 * 4096 / 4);
  // QKV GEMM: (2048 x 6144) = xh(2048x4096) * Wqkv(6144x4096)^T
  gemm_f16_nt_kernel<<<dim3(6144 / 256, 2048 / 64), T, 0, stream>>>(xh, wh, qkv, 4096, 4096, 4096, 6144);
  // RoPE + precision/layout conversion
  rope_q_kernel<<<(2048 * 32 * 64 + T - 1) / T, T, 0, stream>>>(qkv, fc, qh);
  rope_k_kernel<<<(2048 * 8 * 64 + T - 1) / T, T, 0, stream>>>(qkv, fc, kh);
  transpose_v_kernel<<<(8 * 128 * 2048 + T - 1) / T, T, 0, stream>>>(qkv, vt);
  // flash attention: 4096 waves, 8 waves per block
  attn_kernel<<<512, T, 0, stream>>>(qh, kh, vt, xh /* reuse as y (f16) */);
  // output projection
  f32_to_f16_kernel<<<(4096 * 4096 / 4 + T - 1) / T, T, 0, stream>>>(Wo, wh, 4096 * 4096 / 4);
  gemm_f16_nt_kernel<<<dim3(4096 / 256, 2048 / 64), T, 0, stream>>>(xh, wh, out, 4096, 4096, 4096, 4096);
}
